// GPT2ModelBlock_77515569758653
// MI455X (gfx1250) — matrix operand order repaired and checked by reading, not compiled, not run on hardware
//
#include <hip/hip_runtime.h>
#include <hip/hip_bf16.h>
#include <math.h>

// ---------------------------------------------------------------------------
// GPT-2 block on gfx1250 (MI455X). Compute-bound (~470 GFLOP vs ~250MB moved):
// everything routes through v_wmma_f32_16x16x32_bf16 (f32 accumulate).
// Fragments are fed by ds_load_b128 pairs; all transposes are done with
// coalesced b32/b128 loads + v_perm_b32 register packing (no scalar scatter).
// ---------------------------------------------------------------------------

typedef unsigned short u16;
typedef __attribute__((ext_vector_type(16))) __bf16          v16bf;
typedef __attribute__((ext_vector_type(16))) unsigned short  v16u;
typedef __attribute__((ext_vector_type(8)))  unsigned short  v8u;
typedef __attribute__((ext_vector_type(4)))  unsigned int    v4ui;
typedef __attribute__((ext_vector_type(8)))  float           v8f;

#define DEVINL __device__ __forceinline__

DEVINL u16 f2bf(float f) {
  union { float f; unsigned u; } v; v.f = f;
  unsigned u = v.u;
  if ((u & 0x7F800000u) != 0x7F800000u)      // round-to-nearest-even
    u += 0x7FFFu + ((u >> 16) & 1u);
  return (u16)(u >> 16);
}

DEVINL v8f wmma_bf16(v16u a, v16u b, v8f c) {
  // D = A(16x32 bf16) * B(32x16 bf16) + C(16x16 f32)
  return __builtin_amdgcn_wmma_f32_16x16x32_bf16(
      false, __builtin_bit_cast(v16bf, a),
      false, __builtin_bit_cast(v16bf, b),
      (short)0, c, false, false);
}

// v_perm_b32 lane-local 16-bit transpose helpers:
// pack_lo(x,y) = {x.lo16, y.lo16}, pack_hi(x,y) = {x.hi16, y.hi16}
DEVINL unsigned pack_lo(unsigned x, unsigned y) { return __builtin_amdgcn_perm(y, x, 0x05040100u); }
DEVINL unsigned pack_hi(unsigned x, unsigned y) { return __builtin_amdgcn_perm(y, x, 0x07060302u); }

// Build a 16-slot fragment from two 16-byte runs (ds_load_b128 x2).
DEVINL v16u ld_frag(const u16* p0, const u16* p1) {
  v8u lo = *(const v8u*)p0;
  v8u hi = *(const v8u*)p1;
  return __builtin_shufflevector(lo, hi, 0, 1, 2, 3, 4, 5, 6, 7,
                                         8, 9, 10, 11, 12, 13, 14, 15);
}
// A-layout fragment (16x32 16-bit, ISA 7.12.2): slots 0-7 = K half*8..+7,
// slots 8-15 = K 16+half*8..+7 -> base p and p+16 (row-major, K contiguous).
DEVINL v16u ld_afrag(const u16* p) { return ld_frag(p, p + 16); }
// B-layout fragment: same K order as A (slots 0-7 = K half*8..+7, 8-15 = K 16+half*8..+7) -> p and p+16.
DEVINL v16u ld_bfrag(const u16* p) { return ld_frag(p, p + 16); }

// ---------------------------------------------------------------------------
// LayerNorm: fp32 row -> bf16 row
// ---------------------------------------------------------------------------
__global__ __launch_bounds__(256) void k_layernorm_bf16(
    const float* __restrict__ x, const float* __restrict__ g,
    const float* __restrict__ b, u16* __restrict__ out, int E) {
  int row = blockIdx.x;
  const float* xr = x + (size_t)row * E;
  u16* orow = out + (size_t)row * E;
  float s = 0.f, sq = 0.f;
  for (int i = threadIdx.x; i < E; i += 256) { float v = xr[i]; s += v; sq += v * v; }
#pragma unroll
  for (int o = 16; o > 0; o >>= 1) { s += __shfl_xor(s, o, 32); sq += __shfl_xor(sq, o, 32); }
  __shared__ float rs[8], rq[8];
  int wave = threadIdx.x >> 5, lane = threadIdx.x & 31;
  if (lane == 0) { rs[wave] = s; rq[wave] = sq; }
  __syncthreads();
  s = 0.f; sq = 0.f;
#pragma unroll
  for (int i = 0; i < 8; ++i) { s += rs[i]; sq += rq[i]; }
  float mu  = s / (float)E;
  float var = sq / (float)E - mu * mu;
  float inv = rsqrtf(var + 1e-5f);
  for (int i = threadIdx.x; i < E; i += 256)
    orow[i] = f2bf((xr[i] - mu) * inv * g[i] + b[i]);
}

// ---------------------------------------------------------------------------
// fp32 -> bf16 conversion (weights)
// ---------------------------------------------------------------------------
__global__ __launch_bounds__(256) void k_f32_to_bf16(
    const float* __restrict__ in, u16* __restrict__ out, int n) {
  int stride = gridDim.x * 256;
  for (int i = blockIdx.x * 256 + threadIdx.x; i < n; i += stride)
    out[i] = f2bf(in[i]);
}

// ---------------------------------------------------------------------------
// bf16 WMMA GEMM: C(MxN) = A(MxK)*B(KxN) + bias. Block tile 128x64, BK=64.
// 8 waves (4 M x 2 N), 32x32 per wave -> 8 WMMA per stage iteration.
// B staged transposed (Bst[n][k]) via b32 loads + v_perm packing.
// ---------------------------------------------------------------------------
enum { EP_BF16 = 0, EP_QKV = 1, EP_GELU = 2, EP_RESID = 3 };

template <int MODE>
__global__ __launch_bounds__(256) void k_gemm_bf16(
    const u16* __restrict__ A, const u16* __restrict__ Bm,
    const float* __restrict__ bias, const float* __restrict__ resid,
    void* __restrict__ outp, int M, int N, int K, int heads) {
  constexpr int LDA = 72;   // 144B rows: 16B aligned, conflict-free spread
  constexpr int LDBT = 72;
  __shared__ __align__(16) u16 As[128 * LDA];
  __shared__ __align__(16) u16 Bst[64 * LDBT];
  const int bm = blockIdx.y * 128, bn = blockIdx.x * 64;
  const int tid = threadIdx.x;
  const int wave = tid >> 5, lane = tid & 31;
  const int wm = (wave & 3) * 32, wn = (wave >> 2) * 32;
  const int cl = lane & 15, half = lane >> 4;
  v8f acc[2][2] = {};
  const int ar = tid >> 1, ac = (tid & 1) * 32;     // A: 128 rows x 64 k
  const int n2 = (tid & 31) * 2, kx = (tid >> 5) * 8; // Bst: 2 n-cols x 8 k
  const u16* aptr = A + (size_t)(bm + ar) * K + ac;
  for (int k0 = 0; k0 < K; k0 += 64) {
    __syncthreads();
    // stage A (direct b128 copies)
#pragma unroll
    for (int i = 0; i < 32; i += 8)
      *(ulonglong2*)&As[ar * LDA + ac + i] = *(const ulonglong2*)(aptr + i);
    aptr += 64;
    // stage B transposed: 8 coalesced b32 loads + v_perm pack + 2 b128 stores
    {
      const u16* bp = Bm + (size_t)(k0 + kx) * N + bn + n2;
      unsigned xw[8];
#pragma unroll
      for (int j = 0; j < 8; ++j) xw[j] = *(const unsigned*)(bp + (size_t)j * N);
      v4ui lo, hi;
#pragma unroll
      for (int j = 0; j < 4; ++j) {
        lo[j] = pack_lo(xw[2 * j], xw[2 * j + 1]);
        hi[j] = pack_hi(xw[2 * j], xw[2 * j + 1]);
      }
      *(v4ui*)&Bst[n2 * LDBT + kx]       = lo;
      *(v4ui*)&Bst[(n2 + 1) * LDBT + kx] = hi;
    }
    if (k0 + 64 < K) __builtin_prefetch(aptr, 0, 0);
    __syncthreads();
#pragma unroll
    for (int kc = 0; kc < 2; ++kc) {
      const int ko = kc * 32;
      v16u a0 = ld_afrag(&As[(wm + cl) * LDA + ko + half * 8]);
      v16u a1 = ld_afrag(&As[(wm + 16 + cl) * LDA + ko + half * 8]);
      v16u b0 = ld_bfrag(&Bst[(wn + cl) * LDBT + ko + half * 8]);
      v16u b1 = ld_bfrag(&Bst[(wn + 16 + cl) * LDBT + ko + half * 8]);
      acc[0][0] = wmma_bf16(a0, b0, acc[0][0]);
      acc[0][1] = wmma_bf16(a0, b1, acc[0][1]);
      acc[1][0] = wmma_bf16(a1, b0, acc[1][0]);
      acc[1][1] = wmma_bf16(a1, b1, acc[1][1]);
    }
  }
  // epilogue. C layout: VGPR r -> row half*8+r, col = lane&15 (+16 per frag)
#pragma unroll
  for (int f = 0; f < 2; ++f)
#pragma unroll
    for (int r = 0; r < 8; ++r) {
      int gm = bm + wm + f * 16 + half * 8 + r;
#pragma unroll
      for (int j = 0; j < 2; ++j) {
        int gn = bn + wn + j * 16 + cl;
        float v = acc[f][j][r] + bias[gn];
        size_t flat = (size_t)gm * N + gn;
        if (MODE == EP_QKV) {
          // (B*S, heads*128) -> (B, heads, S, 128); S=2048, D=128 fixed
          int b_ = gm >> 11, s_ = gm & 2047;
          int h_ = gn >> 7,  d_ = gn & 127;
          size_t idx = (((size_t)b_ * heads + h_) * 2048 + s_) * 128 + d_;
          ((u16*)outp)[idx] = f2bf(v);
        } else if (MODE == EP_GELU) {
          ((u16*)outp)[flat] = f2bf(0.5f * v * (1.0f + erff(v * 0.70710678118654752f)));
        } else if (MODE == EP_RESID) {
          ((float*)outp)[flat] = resid[flat] + v;
        } else {
          ((u16*)outp)[flat] = f2bf(v);
        }
      }
    }
}

// ---------------------------------------------------------------------------
// Causal GQA flash attention. Block = (b, h, 128-row q tile); 8 waves, each
// owning 16 q rows. Per 64-key step: S = Q K^T (16 WMMA), online softmax,
// O += P V (16 WMMA). Q frags in registers (global b128); K tile row-major
// (already B-layout: B = K^T); V transposed via v_perm packing.
// ---------------------------------------------------------------------------
__global__ __launch_bounds__(256) void k_attention(
    const u16* __restrict__ q, const u16* __restrict__ k,
    const u16* __restrict__ v, u16* __restrict__ o) {
  constexpr int S = 2048, D = 128, H = 16, KH = 4, QT = 128, KT = 64;
  constexpr int LKS = D + 8;    // Ks[key][d]  : 272B rows
  constexpr int LVT = KT + 8;   // Vt[d][key]  : 144B rows
  constexpr int LP  = 40;       // Pw[row][key]: 80B rows
  __shared__ __align__(16) u16 Ks[KT * LKS];
  __shared__ __align__(16) u16 Vt[D * LVT];
  __shared__ __align__(16) u16 Ps[8 * 16 * LP];
  const int tid = threadIdx.x, wave = tid >> 5, lane = tid & 31;
  const int cl = lane & 15, half = lane >> 4;
  const int blk = blockIdx.x;
  const int qt = blk & 15;
  const int h  = (blk >> 4) & (H - 1);
  const int b  = blk >> 8;
  const int kvh = h >> 2;                       // GQA factor 4
  const u16* qbase = q + (((size_t)b * H  + h)   * S) * D;
  const u16* kbase = k + (((size_t)b * KH + kvh) * S) * D;
  const u16* vbase = v + (((size_t)b * KH + kvh) * S) * D;

  // Q fragments (A layout) from global: rows qrow0 + cl, 4 K-chunks of 32
  const int qrow0 = qt * QT + wave * 16;
  const u16* qrow = qbase + (size_t)(qrow0 + cl) * D;
  v16u qf[4];
#pragma unroll
  for (int c = 0; c < 4; ++c)
    qf[c] = ld_afrag(qrow + c * 32 + half * 8);

  v8f of[8] = {};                               // O accum: 16x128 per wave
  float m[8], l[8];
#pragma unroll
  for (int r = 0; r < 8; ++r) { m[r] = -1e30f; l[r] = 0.f; }
  u16* Pw = &Ps[wave * 16 * LP];

  const int ktmax = 2 * qt + 2;                 // causal: keys <= tile max row
  for (int kt = 0; kt < ktmax; ++kt) {
    __syncthreads();
    {   // stage K row-major (b128 copies)
      int key = tid >> 2, d0 = (tid & 3) * 32;
      const u16* kr = kbase + (size_t)(kt * KT + key) * D + d0;
#pragma unroll
      for (int i = 0; i < 32; i += 8)
        *(ulonglong2*)&Ks[key * LKS + d0 + i] = *(const ulonglong2*)(kr + i);
    }
    {   // stage V transposed: 4 keys x 8 d per thread, v_perm 4-key packing
      int key4 = (tid & 15) * 4, dv0 = (tid >> 4) * 8;
      const u16* vr = vbase + (size_t)(kt * KT + key4) * D + dv0;
      v4ui va = __builtin_bit_cast(v4ui, *(const v8u*)vr);
      v4ui vb_ = __builtin_bit_cast(v4ui, *(const v8u*)(vr + D));
      v4ui vc = __builtin_bit_cast(v4ui, *(const v8u*)(vr + 2 * D));
      v4ui vd = __builtin_bit_cast(v4ui, *(const v8u*)(vr + 3 * D));
#pragma unroll
      for (int j = 0; j < 4; ++j) {
        uint2 t0, t1;
        t0.x = pack_lo(va[j], vb_[j]); t0.y = pack_lo(vc[j], vd[j]);
        t1.x = pack_hi(va[j], vb_[j]); t1.y = pack_hi(vc[j], vd[j]);
        *(uint2*)&Vt[(dv0 + 2 * j) * LVT + key4]     = t0;
        *(uint2*)&Vt[(dv0 + 2 * j + 1) * LVT + key4] = t1;
      }
    }
    __syncthreads();
    // scores: B-frag = b128 pair from one Ks row (B = K^T)
    v8f sc[4] = {};
#pragma unroll
    for (int c = 0; c < 4; ++c)
#pragma unroll
      for (int nf = 0; nf < 4; ++nf)
        sc[nf] = wmma_bf16(qf[c],
                           ld_bfrag(&Ks[(nf * 16 + cl) * LKS + c * 32 + half * 8]),
                           sc[nf]);
    // scale + causal mask
    const float scale = 0.088388347648318447f;  // 1/sqrt(128)
#pragma unroll
    for (int nf = 0; nf < 4; ++nf)
#pragma unroll
      for (int r = 0; r < 8; ++r) {
        int col  = kt * KT + nf * 16 + cl;
        int rowg = qrow0 + half * 8 + r;
        sc[nf][r] = (col <= rowg) ? sc[nf][r] * scale : -1e30f;
      }
    // online softmax (16-lane row reductions; lanes 0-15 rows 0-7, 16-31 rows 8-15)
#pragma unroll
    for (int r = 0; r < 8; ++r) {
      float mx = fmaxf(fmaxf(sc[0][r], sc[1][r]), fmaxf(sc[2][r], sc[3][r]));
#pragma unroll
      for (int off = 1; off < 16; off <<= 1) mx = fmaxf(mx, __shfl_xor(mx, off, 32));
      float mn = fmaxf(m[r], mx);
      float corr = __expf(m[r] - mn);
      m[r] = mn;
      l[r] *= corr;
#pragma unroll
      for (int d = 0; d < 8; ++d) of[d][r] *= corr;
      float rsum = 0.f;
#pragma unroll
      for (int nf = 0; nf < 4; ++nf) {
        float e = __expf(sc[nf][r] - mn);
        sc[nf][r] = e;
        rsum += e;
      }
#pragma unroll
      for (int off = 1; off < 16; off <<= 1) rsum += __shfl_xor(rsum, off, 32);
      l[r] += rsum;
    }
    // P (C layout) -> per-wave LDS row-major bf16
#pragma unroll
    for (int nf = 0; nf < 4; ++nf)
#pragma unroll
      for (int r = 0; r < 8; ++r)
        Pw[(half * 8 + r) * LP + nf * 16 + cl] = f2bf(sc[nf][r]);
    // O += P @ V  (pa: A-frag b128 pair; vb: b128 pair from one Vt row)
#pragma unroll
    for (int kc = 0; kc < 2; ++kc) {
      v16u pa = ld_afrag(&Pw[cl * LP + kc * 32 + half * 8]);
#pragma unroll
      for (int d = 0; d < 8; ++d)
        of[d] = wmma_bf16(pa,
                          ld_bfrag(&Vt[(d * 16 + cl) * LVT + kc * 32 + half * 8]),
                          of[d]);
    }
  }
  // normalize + store into o[(b,s,h*128+d)] (GEMM-ready row major)
#pragma unroll
  for (int r = 0; r < 8; ++r) {
    float inv = 1.0f / l[r];
    int srow = qrow0 + half * 8 + r;
    u16* orow = o + ((size_t)b * S + srow) * (H * D) + h * D;
#pragma unroll
    for (int d = 0; d < 8; ++d)
      orow[d * 16 + cl] = f2bf(of[d][r] * inv);
  }
}

// ---------------------------------------------------------------------------
// Host orchestration
// ---------------------------------------------------------------------------
extern "C" void kernel_launch(void* const* d_in, const int* in_sizes, int n_in,
                              void* d_out, int out_size, void* d_ws, size_t ws_size,
                              hipStream_t stream) {
  (void)in_sizes; (void)n_in; (void)out_size; (void)ws_size;
  constexpr int B = 2, S = 2048, E = 2048, H = 16, D = 128, KH = 4, F = 8192;
  constexpr int Mr = B * S;  // 4096 token rows

  const float* x    = (const float*)d_in[0];
  const float* ln1g = (const float*)d_in[1];
  const float* ln1b = (const float*)d_in[2];
  const float* wq   = (const float*)d_in[3];
  const float* bq   = (const float*)d_in[4];
  const float* wk   = (const float*)d_in[5];
  const float* bk   = (const float*)d_in[6];
  const float* wv   = (const float*)d_in[7];
  const float* bv   = (const float*)d_in[8];
  const float* wo   = (const float*)d_in[9];
  const float* bo   = (const float*)d_in[10];
  const float* ln2g = (const float*)d_in[11];
  const float* ln2b = (const float*)d_in[12];
  const float* wu   = (const float*)d_in[13];
  const float* bu   = (const float*)d_in[14];
  const float* wd   = (const float*)d_in[15];
  const float* bd   = (const float*)d_in[16];
  float* out = (float*)d_out;

  char* ws = (char*)d_ws;
  size_t off = 0;
  auto alloc = [&](size_t bytes) -> void* {
    void* p = ws + off; off += (bytes + 255) & ~(size_t)255; return p;
  };
  u16* xln  = (u16*)alloc((size_t)Mr * E * 2);        // LN output (reused for LN2)
  u16* wbuf = (u16*)alloc((size_t)E * F * 2);         // shared bf16 weight buffer
  u16* qb   = (u16*)alloc((size_t)B * H  * S * D * 2);
  u16* kb   = (u16*)alloc((size_t)B * KH * S * D * 2);
  u16* vb   = (u16*)alloc((size_t)B * KH * S * D * 2);
  u16* ob   = (u16*)alloc((size_t)Mr * H * D * 2);
  u16* hb   = (u16*)alloc((size_t)Mr * F * 2);

  dim3 blk(256);
  auto cvt = [&](const float* src, u16* dst, size_t n) {
    int g = (int)((n + 2047) / 2048);
    k_f32_to_bf16<<<g, blk, 0, stream>>>(src, dst, (int)n);
  };

  // LN1
  k_layernorm_bf16<<<Mr, blk, 0, stream>>>(x, ln1g, ln1b, xln, E);
  // Q/K/V projections (head-permuted bf16 outputs)
  cvt(wq, wbuf, (size_t)E * H * D);
  k_gemm_bf16<EP_QKV><<<dim3((H * D) / 64, Mr / 128), blk, 0, stream>>>(
      xln, wbuf, bq, nullptr, qb, Mr, H * D, E, H);
  cvt(wk, wbuf, (size_t)E * KH * D);
  k_gemm_bf16<EP_QKV><<<dim3((KH * D) / 64, Mr / 128), blk, 0, stream>>>(
      xln, wbuf, bk, nullptr, kb, Mr, KH * D, E, KH);
  cvt(wv, wbuf, (size_t)E * KH * D);
  k_gemm_bf16<EP_QKV><<<dim3((KH * D) / 64, Mr / 128), blk, 0, stream>>>(
      xln, wbuf, bv, nullptr, vb, Mr, KH * D, E, KH);
  // causal GQA attention
  k_attention<<<B * H * (S / 128), blk, 0, stream>>>(qb, kb, vb, ob);
  // output projection + residual (fp32 into d_out)
  cvt(wo, wbuf, (size_t)(H * D) * E);
  k_gemm_bf16<EP_RESID><<<dim3(E / 64, Mr / 128), blk, 0, stream>>>(
      ob, wbuf, bo, x, out, Mr, E, H * D, 0);
  // LN2
  k_layernorm_bf16<<<Mr, blk, 0, stream>>>(out, ln2g, ln2b, xln, E);
  // MLP up + exact GELU
  cvt(wu, wbuf, (size_t)E * F);
  k_gemm_bf16<EP_GELU><<<dim3(F / 64, Mr / 128), blk, 0, stream>>>(
      xln, wbuf, bu, nullptr, hb, Mr, F, E, 0);
  // MLP down + residual (in-place on d_out: same-thread read-then-write)
  cvt(wd, wbuf, (size_t)F * E);
  k_gemm_bf16<EP_RESID><<<dim3(E / 64, Mr / 128), blk, 0, stream>>>(
      hb, wbuf, bd, out, out, Mr, E, F, 0);
}
